// HybridBlock_31533649887822
// MI455X (gfx1250) — matrix operand order repaired and checked by reading, not compiled, not run on hardware
//
#include <hip/hip_runtime.h>

// ---------------- problem constants (from reference) ----------------
#define NN      40000
#define EE      640000
#define IN_DIM  128
#define EDGE_DIM 64
#define HID     128
#define KTOT    320            // EDGE_DIM + 2*IN_DIM
#define KT      10             // 320 / 32
#define NT      8              // 128 / 16
#define RED_BLOCKS 1024

typedef unsigned int u32;
typedef __attribute__((ext_vector_type(4)))  u32    u32x4;
typedef __attribute__((ext_vector_type(8)))  int    i32x8;
typedef __attribute__((ext_vector_type(4)))  int    i32x4;
typedef __attribute__((ext_vector_type(16))) __bf16 bf16x16;
typedef __attribute__((ext_vector_type(8)))  float  f32x8;
typedef __attribute__((ext_vector_type(2)))  float  f32x2;

union ABfrag { u32x4 u[2]; bf16x16 v; };
union PK2    { __bf16 h[2]; u32 u; };

#if defined(__has_builtin)
#if __has_builtin(__builtin_amdgcn_tensor_load_to_lds)
#define HAVE_TDM 1
#endif
#endif

// ------------------------------------------------------------------
// K0: pack concat(Wh_w[64x128], Wn_w[256x128]) into bf16 WMMA-B
// fragments: wsB[(kt*8+nt)*32 + lane][0..7] dwords, contiguous per lane.
// B k order used here (32x16 bf16, same as A): lanes 0-15 hold K=0..15 (2/VGPR), lanes 16-31 K=16..31.
__global__ void k_pack_weights(const float* __restrict__ Whw,
                               const float* __restrict__ Wnw,
                               u32* __restrict__ wsB) {
    const int frag = blockIdx.x;          // 0..79  (kt*8+nt)
    const int kt = frag >> 3, nt = frag & 7;
    const int lane = threadIdx.x;         // 0..31
    const int n = nt * 16 + (lane & 15);
    const int kbase = kt * 32 + (lane >> 4) * 16;
    u32* dst = wsB + ((size_t)(frag * 32 + lane)) * 8;
    for (int dw = 0; dw < 8; ++dw) {
        int k0 = kbase + 2 * dw;
        int k1 = k0 + 1;
        float w0 = (k0 < EDGE_DIM) ? Whw[k0 * HID + n] : Wnw[(k0 - EDGE_DIM) * HID + n];
        float w1 = (k1 < EDGE_DIM) ? Whw[k1 * HID + n] : Wnw[(k1 - EDGE_DIM) * HID + n];
        PK2 p; p.h[0] = (__bf16)w0; p.h[1] = (__bf16)w1;
        dst[dw] = p.u;
    }
}

// ------------------------------------------------------------------
// K1: zero scatter buffer (graph replay => must re-zero every launch)
__global__ void k_zero(float4* __restrict__ p, int n4) {
    for (int i = blockIdx.x * blockDim.x + threadIdx.x; i < n4;
         i += gridDim.x * blockDim.x)
        p[i] = make_float4(0.f, 0.f, 0.f, 0.f);
}

// ------------------------------------------------------------------
// K2: per-edge scores via bf16 WMMA.
// 128 threads = 4 waves; each wave owns a tile of 16 edges.
#define FSTRIDE 336   // 320 + 16 bf16 pad (keeps 16B alignment: 336%8==0)
__global__ void k_scores(const float* __restrict__ x,
                         const long long* __restrict__ ei,
                         const float* __restrict__ ea,
                         const float* __restrict__ Whb,
                         const float* __restrict__ Wnb,
                         const float* __restrict__ we,
                         const u32* __restrict__ wsB,
                         float* __restrict__ scores) {
    __shared__ __bf16 feat[4][16][FSTRIDE];
    const int wave = threadIdx.x >> 5;
    const int lane = threadIdx.x & 31;
    const int tile = blockIdx.x * 4 + wave;     // 0..39999
    const int e0 = tile * 16;

    // ---- stage gathered features for 16 edges as bf16 rows of K=320 ----
    for (int e = 0; e < 16; ++e) {
        const int eid = e0 + e;
        const long long s = ei[eid];
        const long long d = ei[EE + eid];
        // edge_attr: 64 floats, 2 per lane
        {
            int c = lane * 2;
            const float2 v = *(const float2*)(ea + (size_t)eid * EDGE_DIM + c);
            feat[wave][e][c]     = (__bf16)v.x;
            feat[wave][e][c + 1] = (__bf16)v.y;
        }
        // x[src]: 128 floats, 4 per lane
        {
            const float4 v = ((const float4*)(x + (size_t)s * IN_DIM))[lane];
            int c = EDGE_DIM + lane * 4;
            feat[wave][e][c]     = (__bf16)v.x;
            feat[wave][e][c + 1] = (__bf16)v.y;
            feat[wave][e][c + 2] = (__bf16)v.z;
            feat[wave][e][c + 3] = (__bf16)v.w;
        }
        // x[dst]
        {
            const float4 v = ((const float4*)(x + (size_t)d * IN_DIM))[lane];
            int c = EDGE_DIM + IN_DIM + lane * 4;
            feat[wave][e][c]     = (__bf16)v.x;
            feat[wave][e][c + 1] = (__bf16)v.y;
            feat[wave][e][c + 2] = (__bf16)v.z;
            feat[wave][e][c + 3] = (__bf16)v.w;
        }
    }
    __syncthreads();

    // ---- epilogue constants per lane ----
    const int nl = lane & 15;
    float bias[NT], wev[NT];
    for (int nt = 0; nt < NT; ++nt) {
        int n = nt * 16 + nl;
        bias[nt] = Whb[n] + Wnb[n];
        wev[nt]  = we[n];
    }

    // ---- WMMA main loop: K=320 in 10 steps of 32, 8 N-tiles ----
    f32x8 acc[NT] = {};
    const int r  = lane & 15;
    for (int kt = 0; kt < KT; ++kt) {
        const int kb = kt * 32 + (lane >> 4) * 16;  // A k order used here (same as B): halves hold K+0..15 / K+16..31
        ABfrag a;
        a.u[0] = *(const u32x4*)&feat[wave][r][kb];
        a.u[1] = *(const u32x4*)&feat[wave][r][kb + 8];
        for (int nt = 0; nt < NT; ++nt) {
            const u32x4* q = (const u32x4*)(wsB + ((size_t)((kt * NT + nt) * 32 + lane)) * 8);
            ABfrag b;
            b.u[0] = q[0];
            b.u[1] = q[1];
            acc[nt] = __builtin_amdgcn_wmma_f32_16x16x32_bf16(
                false, a.v, false, b.v, (short)0, acc[nt], false, false);
        }
    }

    // ---- score = sum_n relu(h + bias) * w_e ----
    float s[8];
    for (int j = 0; j < 8; ++j) s[j] = 0.f;
    for (int nt = 0; nt < NT; ++nt)
        for (int j = 0; j < 8; ++j)
            s[j] += fmaxf(acc[nt][j] + bias[nt], 0.f) * wev[nt];
    // reduce across the 16 lanes sharing each row half
    for (int off = 1; off < 16; off <<= 1)
        for (int j = 0; j < 8; ++j)
            s[j] += __shfl_xor(s[j], off, 32);
    if ((lane & 15) == 0) {
        const int mb = (lane >> 4) * 8;   // lanes 0-15 -> rows 0..7, 16-31 -> rows 8..15
        for (int j = 0; j < 8; ++j)
            scores[e0 + mb + j] = s[j];
    }
}

// ------------------------------------------------------------------
// K3/K4: softmax max reduction (two level)
__global__ void k_max_partial(const float* __restrict__ sc, float* __restrict__ pmax) {
    __shared__ float sm[256];
    float m = -3.4e38f;
    for (int i = blockIdx.x * 256 + threadIdx.x; i < EE; i += 256 * gridDim.x)
        m = fmaxf(m, sc[i]);
    sm[threadIdx.x] = m; __syncthreads();
    for (int s = 128; s > 0; s >>= 1) {
        if (threadIdx.x < s) sm[threadIdx.x] = fmaxf(sm[threadIdx.x], sm[threadIdx.x + s]);
        __syncthreads();
    }
    if (threadIdx.x == 0) pmax[blockIdx.x] = sm[0];
}
__global__ void k_max_final(const float* __restrict__ pmax, float* __restrict__ g) {
    __shared__ float sm[256];
    float m = -3.4e38f;
    for (int i = threadIdx.x; i < RED_BLOCKS; i += 256) m = fmaxf(m, pmax[i]);
    sm[threadIdx.x] = m; __syncthreads();
    for (int s = 128; s > 0; s >>= 1) {
        if (threadIdx.x < s) sm[threadIdx.x] = fmaxf(sm[threadIdx.x], sm[threadIdx.x + s]);
        __syncthreads();
    }
    if (threadIdx.x == 0) g[0] = sm[0];
}

// K5/K6: exp-sum reduction
__global__ void k_sum_partial(const float* __restrict__ sc, const float* __restrict__ g,
                              float* __restrict__ psum) {
    __shared__ float sm[256];
    const float gm = g[0];
    float a = 0.f;
    for (int i = blockIdx.x * 256 + threadIdx.x; i < EE; i += 256 * gridDim.x)
        a += __expf(sc[i] - gm);
    sm[threadIdx.x] = a; __syncthreads();
    for (int s = 128; s > 0; s >>= 1) {
        if (threadIdx.x < s) sm[threadIdx.x] += sm[threadIdx.x + s];
        __syncthreads();
    }
    if (threadIdx.x == 0) psum[blockIdx.x] = sm[0];
}
__global__ void k_sum_final(const float* __restrict__ psum, float* __restrict__ g) {
    __shared__ float sm[256];
    float a = 0.f;
    for (int i = threadIdx.x; i < RED_BLOCKS; i += 256) a += psum[i];
    sm[threadIdx.x] = a; __syncthreads();
    for (int s = 128; s > 0; s >>= 1) {
        if (threadIdx.x < s) sm[threadIdx.x] += sm[threadIdx.x + s];
        __syncthreads();
    }
    if (threadIdx.x == 0) g[1] = sm[0];
}

// ------------------------------------------------------------------
// K7: scatter  local[src] += -alpha * x[dst]   (8 edges / block, 1 wave / edge)
__global__ void k_scatter(const float* __restrict__ x,
                          const long long* __restrict__ ei,
                          const float* __restrict__ sc,
                          const float* __restrict__ g,
                          float* __restrict__ local) {
    const int eid  = blockIdx.x * 8 + (threadIdx.x >> 5);
    const int lane = threadIdx.x & 31;
    const float alpha = __expf(sc[eid] - g[0]) / g[1];
    const long long s = ei[eid];
    const long long d = ei[EE + eid];
    const float4 xd = ((const float4*)(x + (size_t)d * IN_DIM))[lane];
    float* lp = local + (size_t)s * IN_DIM + lane * 4;
    atomicAdd(lp + 0, -alpha * xd.x);
    atomicAdd(lp + 1, -alpha * xd.y);
    atomicAdd(lp + 2, -alpha * xd.z);
    atomicAdd(lp + 3, -alpha * xd.w);
}

// ------------------------------------------------------------------
// K8: out = t + t@ft_w + ft_b, t = x + local, fp32 WMMA 16x16x4.
// 128 threads = 4 waves; block (bx, by) owns nodes [bx*16,..) x cols [by*64,..).
// ft_w column half (128 rows x 64 cols, fp32) is staged into LDS by the
// Tensor Data Mover (tensor_load_to_lds + s_wait_tensorcnt), so the WMMA
// inner loop runs entirely out of LDS.
__global__ void k_final(const float* __restrict__ x,
                        const float* __restrict__ local,
                        const float* __restrict__ ftw,
                        const float* __restrict__ ftb,
                        float* __restrict__ out) {
    __shared__ float t[16][132];
    __shared__ float wlds[IN_DIM * 64];          // 128 rows x 64 cols, row stride 64
    const int n0   = blockIdx.x * 16;
    const int coff = blockIdx.y * 64;

#if HAVE_TDM
    if (threadIdx.x < 32) {   // wave 0 issues the TDM op (EXEC is ignored by TDM)
        const u32 ldsoff = (u32)(size_t)(&wlds[0]);  // low 32 bits == LDS offset
        const unsigned long long ga =
            (unsigned long long)(size_t)(ftw + (size_t)coff);
        // D# group 0: count=1 | lds_addr | global_addr[56:0] | type=2
        u32x4 g0 = { 1u, ldsoff, (u32)ga,
                     (u32)((ga >> 32) & 0x01FFFFFFu) | (2u << 30) };
        // D# group 1: data_size=4B; tensor 128x128; tile 64x128; stride 128
        i32x8 g1 = { (int)0x00020000u,          // wg_mask=0, data_size=2 (4B)
                     (int)(128u << 16),          // tensor_dim0 = 128
                     (int)(128u << 16),          // tensor_dim1 = 128
                     (int)(64u  << 16),          // tile_dim0   = 64
                     128,                        // tile_dim1   = 128
                     128,                        // tensor_dim0_stride = 128
                     0, 0 };
        i32x4 g2 = { 0, 0, 0, 0 };
        i32x4 g3 = { 0, 0, 0, 0 };
        i32x8 g4 = { 0, 0, 0, 0, 0, 0, 0, 0 };   // clang-23 6-arg form: extra group
        __builtin_amdgcn_tensor_load_to_lds(g0, g1, g2, g3, g4, 0);
        __builtin_amdgcn_s_wait_tensorcnt(0);
    }
#else
    for (int i = threadIdx.x; i < IN_DIM * 64; i += 128) {
        int rr = i >> 6, cc = i & 63;
        wlds[i] = ftw[(size_t)rr * IN_DIM + coff + cc];
    }
#endif

    // stage t = x + local for 16 nodes (full K=128 needed for the A matrix)
    for (int i = threadIdx.x; i < 16 * IN_DIM; i += 128) {
        int r = i >> 7, c = i & 127;
        size_t idx = (size_t)(n0 + r) * IN_DIM + c;
        t[r][c] = x[idx] + local[idx];
    }
    __syncthreads();

    const int ntw  = threadIdx.x >> 5;        // wave id == column tile (0..3)
    const int lane = threadIdx.x & 31;
    const int nl   = lane & 15;
    const int colL = ntw * 16 + nl;           // 0..63 within the half
    const int col  = coff + colL;
    const int r    = lane & 15;
    f32x8 acc = {};
    for (int ks = 0; ks < 32; ++ks) {
        const int kb = ks * 4 + (lane >> 4) * 2;  // A/B halves hold K+0,1 / K+2,3
        f32x2 a, b;
        a[0] = t[r][kb];
        a[1] = t[r][kb + 1];
        b[0] = wlds[kb * 64 + colL];
        b[1] = wlds[(kb + 1) * 64 + colL];
        acc = __builtin_amdgcn_wmma_f32_16x16x4_f32(
            false, a, false, b, (short)0, acc, false, false);
    }
    const int rb = (lane >> 4) * 8;
    const float bcol = ftb[col];
    for (int j = 0; j < 8; ++j) {
        int row = rb + j;
        out[(size_t)(n0 + row) * IN_DIM + col] = t[row][col] + acc[j] + bcol;
    }
}

// ------------------------------------------------------------------
extern "C" void kernel_launch(void* const* d_in, const int* in_sizes, int n_in,
                              void* d_out, int out_size, void* d_ws, size_t ws_size,
                              hipStream_t stream) {
    const float*     x    = (const float*)d_in[0];
    const long long* ei   = (const long long*)d_in[1];   // int64 edge_index [2,E]
    const float*     ea   = (const float*)d_in[2];
    const float*     Whw  = (const float*)d_in[3];
    const float*     Whb  = (const float*)d_in[4];
    const float*     Wnw  = (const float*)d_in[5];
    const float*     Wnb  = (const float*)d_in[6];
    const float*     we   = (const float*)d_in[7];
    const float*     ftw  = (const float*)d_in[8];
    const float*     ftb  = (const float*)d_in[9];
    float*           out  = (float*)d_out;

    // workspace layout (dwords)
    float* ws     = (float*)d_ws;
    u32*   wsB    = (u32*)ws;                 // 80*32*8      = 20480 dwords
    float* scores = ws + 20480;               // EE           = 640000
    float* pmax   = scores + EE;              // 1024
    float* psum   = pmax + RED_BLOCKS;        // 1024
    float* g      = psum + RED_BLOCKS;        // [gmax, gsum] (+pad to 16 dwords)
    float* local  = g + 16;                   // NN*IN_DIM = 5,120,000

    k_pack_weights<<<80, 32, 0, stream>>>(Whw, Wnw, wsB);
    k_zero<<<2048, 256, 0, stream>>>((float4*)local, (NN * IN_DIM) / 4);
    k_scores<<<EE / 64, 128, 0, stream>>>(x, ei, ea, Whb, Wnb, we, wsB, scores);
    k_max_partial<<<RED_BLOCKS, 256, 0, stream>>>(scores, pmax);
    k_max_final<<<1, 256, 0, stream>>>(pmax, g);
    k_sum_partial<<<RED_BLOCKS, 256, 0, stream>>>(scores, g, psum);
    k_sum_final<<<1, 256, 0, stream>>>(psum, g);
    k_scatter<<<EE / 8, 256, 0, stream>>>(x, ei, scores, g, local);
    k_final<<<dim3(NN / 16, 2, 1), 128, 0, stream>>>(x, local, ftw, ftb, out);
}
